// TransformerDecoderLayer_24446953849401
// MI455X (gfx1250) — matrix operand order repaired and checked by reading, not compiled, not run on hardware
//
#include <hip/hip_runtime.h>

// ---------------- types / constants ----------------
typedef unsigned short u16;
typedef unsigned int   u32;
typedef __attribute__((ext_vector_type(16))) __bf16 v16bf;
typedef __attribute__((ext_vector_type(16))) u16    v16u;
typedef __attribute__((ext_vector_type(8)))  u16    v8u;
typedef __attribute__((ext_vector_type(8)))  float  v8f;

#define BB   16
#define LL   300
#define EE   256
#define HH   8
#define NTOK (BB*LL)     // 4800
#define NPAD 4864        // 76 * 64
#define FFD  1024
#define SCALE_F 0.17677669529663687f   // 1/sqrt(32)

#define V8F_ZERO {0.f,0.f,0.f,0.f,0.f,0.f,0.f,0.f}
#define WMMA_BF16(a,b,c) __builtin_amdgcn_wmma_f32_16x16x32_bf16(false,(a),false,(b),(short)0,(c),false,false)

// fp32 <-> bf16 (round-to-nearest-even), integer path only
__device__ __forceinline__ u16 f2b(float f){
  u32 u = __float_as_uint(f);
  u32 r = u + 0x7FFFu + ((u >> 16) & 1u);
  return (u16)(r >> 16);
}
__device__ __forceinline__ float b2f(u16 h){ return __uint_as_float(((u32)h) << 16); }

// LDS byte offset of a __shared__ object: flat address low 32 bits
__device__ __forceinline__ u32 lds_off(const void* p){ return (u32)(size_t)p; }

// CDNA5 async global->LDS copy, 16B per lane, tracked by ASYNCcnt
__device__ __forceinline__ void async_b128(u32 lds_addr, const void* gptr){
  asm volatile("global_load_async_to_lds_b128 %0, %1, off"
               :: "v"(lds_addr), "v"((unsigned long long)(size_t)gptr)
               : "memory");
}
__device__ __forceinline__ void wait_async0(){
  asm volatile("s_wait_asynccnt 0x0" ::: "memory");
}

// A fragment (16x32 bf16, MxK): lane half=lane/16 supplies K chunks [8*half,+8) and [16+8*half,+8)
__device__ __forceinline__ v16bf make_a_frag(const u16* rowptr, int half){
  v8u c1 = *(const v8u*)(rowptr + 8*half);
  v8u c2 = *(const v8u*)(rowptr + 16 + 8*half);
  v16u a;
#pragma unroll
  for (int i = 0; i < 8; i++){ a[i] = c1[i]; a[i+8] = c2[i]; }
  return __builtin_bit_cast(v16bf, a);
}
// B fragment (32x16 bf16, KxN): lane holds column n=lane%16, K chunks [8*half,+8) and [16+8*half,+8)
__device__ __forceinline__ v16bf make_b_frag(const u16* colptr, int half){
  v8u c1 = *(const v8u*)(colptr + 8*half);
  v8u c2 = *(const v8u*)(colptr + 16 + 8*half);
  v16u b;
#pragma unroll
  for (int i = 0; i < 8; i++){ b[i] = c1[i]; b[i+8] = c2[i]; }
  return __builtin_bit_cast(v16bf, b);
}

// ---------------- elementwise helpers ----------------
__global__ void cvtbf_kernel(const float* __restrict__ a, u16* __restrict__ o, int n){
  int i = blockIdx.x*256 + threadIdx.x;
  if (i < n) o[i] = f2b(a[i]);
}
__global__ void add2bf_kernel(const float* __restrict__ a, const float* __restrict__ b,
                              u16* __restrict__ o, int n){
  int i = blockIdx.x*256 + threadIdx.x;
  if (i < n) o[i] = f2b(a[i] + b[i]);
}
// srcs (B,E,64,64) -> bf16 src (B,64,64,E)
__global__ void src_tr_kernel(const float* __restrict__ srcs, u16* __restrict__ o){
  size_t idx = (size_t)blockIdx.x*256 + threadIdx.x;   // B*4096*256
  if (idx >= (size_t)BB*4096*EE) return;
  int e  = (int)(idx & 255);
  size_t yx = (idx >> 8) & 4095;
  size_t b  = idx >> 20;
  o[idx] = f2b(srcs[(b*EE + e)*4096 + yx]);
}
// mode 0: k_row input = mean_y src + posemb_row ; mode 1: k_col input = mean_x src + posemb_col
__global__ void make_kin_kernel(const float* __restrict__ srcs, const float* __restrict__ pos,
                                u16* __restrict__ o, int mode){
  int idx = blockIdx.x*256 + threadIdx.x;              // B*64*256
  if (idx >= BB*64*EE) return;
  int e = idx & 255, p = (idx >> 8) & 63, b = idx >> 14;
  const float* base = srcs + ((size_t)b*EE + e)*4096;
  float s = 0.f;
  if (mode == 0){ for (int y = 0; y < 64; y++) s += base[y*64 + p]; }
  else          { for (int x = 0; x < 64; x++) s += base[p*64 + x]; }
  float v = s*(1.f/64.f) + pos[((size_t)b*64 + p)*EE + e];
  o[((size_t)b*64 + p)*EE + e] = f2b(v);
}
// out = LN(X + D) * w + b ; optional bf16 copy
__global__ __launch_bounds__(256) void add_ln_kernel(
    const float* __restrict__ X, const float* __restrict__ D,
    const float* __restrict__ w, const float* __restrict__ bvec,
    float* __restrict__ out, u16* __restrict__ outb, int rows){
  int row = blockIdx.x*8 + (threadIdx.x >> 5);
  if (row >= rows) return;
  int lane = threadIdx.x & 31;
  float v[8]; float s = 0.f;
#pragma unroll
  for (int i = 0; i < 8; i++){
    int c = lane + 32*i;
    float t = X[(size_t)row*EE + c] + D[(size_t)row*EE + c];
    v[i] = t; s += t;
  }
#pragma unroll
  for (int off = 16; off > 0; off >>= 1) s += __shfl_xor(s, off, 32);
  float mean = s * (1.f/EE);
  float q = 0.f;
#pragma unroll
  for (int i = 0; i < 8; i++){ float d = v[i]-mean; q += d*d; }
#pragma unroll
  for (int off = 16; off > 0; off >>= 1) q += __shfl_xor(q, off, 32);
  float rstd = rsqrtf(q*(1.f/EE) + 1e-5f);
#pragma unroll
  for (int i = 0; i < 8; i++){
    int c = lane + 32*i;
    float o = (v[i]-mean)*rstd*w[c] + bvec[c];
    out[(size_t)row*EE + c] = o;
    if (outb) outb[(size_t)row*EE + c] = f2b(o);
  }
}

// ---------------- generic bf16 WMMA GEMM: C[M,N] = A[M,K] @ W[N,K]^T + bias ----------------
// block = 64x64 tile, 8 waves (4 M x 2 N), each wave 16x32 via two accumulators, K-step 32.
// Tiles are staged with CDNA5 async global->LDS copies, double-buffered on ASYNCcnt.
// Optional epilogues: fp32 C, bf16 Cb (row-major), bf16 CbT in (b, e, y*64+x) layout for vv.
__global__ __launch_bounds__(256) void gemm_bf16_kernel(
    const u16* __restrict__ A, const u16* __restrict__ W,
    const float* __restrict__ bias, float* __restrict__ C,
    u16* __restrict__ Cb, u16* __restrict__ CbT, int N, int K, int relu){
  __shared__ __align__(32) u16 At[2][64][32];
  __shared__ __align__(32) u16 Wt[2][64][32];
  const int m0 = blockIdx.x*64, n0 = blockIdx.y*64;
  const int tid = threadIdx.x, wave = tid >> 5, lane = tid & 31;
  const int wm = wave & 3, wn = wave >> 2;
  const int half = lane >> 4, l16 = lane & 15;
  const int srow = tid >> 2, sseg = tid & 3;

  const u16* gA = A + (size_t)(m0+srow)*K + sseg*8;
  const u16* gW = W + (size_t)(n0+srow)*K + sseg*8;
  const u32 la[2] = { lds_off(&At[0][srow][sseg*8]), lds_off(&At[1][srow][sseg*8]) };
  const u32 lw[2] = { lds_off(&Wt[0][srow][sseg*8]), lds_off(&Wt[1][srow][sseg*8]) };

  async_b128(la[0], gA);
  async_b128(lw[0], gW);
  wait_async0();
  __syncthreads();

  v8f acc0 = V8F_ZERO, acc1 = V8F_ZERO;
  int cur = 0;
  for (int k0 = 0; k0 < K; k0 += 32){
    int nxt = cur ^ 1;
    if (k0 + 32 < K){                       // prefetch next tile into the other buffer
      async_b128(la[nxt], gA + k0 + 32);
      async_b128(lw[nxt], gW + k0 + 32);
    }
    v16bf a  = make_a_frag(&At[cur][wm*16 + l16][0], half);
    v16bf b0 = make_b_frag(&Wt[cur][wn*32 + l16][0], half);
    v16bf b1 = make_b_frag(&Wt[cur][wn*32 + 16 + l16][0], half);
    acc0 = WMMA_BF16(a, b0, acc0);
    acc1 = WMMA_BF16(a, b1, acc1);
    wait_async0();                          // my next-tile copies have landed
    __syncthreads();                        // everyone done computing & copying
    cur = nxt;
  }
#pragma unroll
  for (int r = 0; r < 8; r++){
    int rg  = m0 + wm*16 + r + 8*half;
    int c0g = n0 + wn*32 + l16, c1g = c0g + 16;
    float v0 = acc0[r] + (bias ? bias[c0g] : 0.f);
    float v1 = acc1[r] + (bias ? bias[c1g] : 0.f);
    if (relu){ v0 = fmaxf(v0, 0.f); v1 = fmaxf(v1, 0.f); }
    size_t base = (size_t)rg * N;
    if (C ){ C [base + c0g] = v0;      C [base + c1g] = v1; }
    if (Cb){ Cb[base + c0g] = f2b(v0); Cb[base + c1g] = f2b(v1); }
    if (CbT){ // (b, e, y*64+x): row rg = b*4096 + yx, col = e
      size_t tb = (size_t)(rg >> 12) * (256*4096) + (size_t)(rg & 4095);
      CbT[tb + (size_t)c0g*4096] = f2b(v0);
      CbT[tb + (size_t)c1g*4096] = f2b(v1);
    }
  }
}

// ---------------- self-attention: P = softmax(scale * q k^T), P bf16 [B,H,300,320] ----------------
__global__ __launch_bounds__(256) void attn_scores_kernel(
    const u16* __restrict__ qk, u16* __restrict__ P){
  const int bh = blockIdx.x, b = bh >> 3, hh = bh & 7;
  const int lt = blockIdx.y;                       // 16 query rows per block
  const int tid = threadIdx.x, wave = tid >> 5, lane = tid & 31;
  const int half = lane >> 4, l16 = lane & 15;
  __shared__ __align__(16) float S[16][320];
  __shared__ float red[16][17];
  {
    int l = lt*16 + l16; if (l > LL-1) l = LL-1;   // clamp; pad rows discarded later
    const u16* qrow = qk + (size_t)(b*LL + l)*512 + hh*32;
    v16bf a = make_a_frag(qrow, half);
    for (int ct = wave; ct < 19; ct += 8){         // wave-uniform key-tile loop
      int key = ct*16 + l16; if (key > LL-1) key = LL-1;
      const u16* krow = qk + (size_t)(b*LL + key)*512 + 256 + hh*32;
      v16bf bf = make_b_frag(krow, half);
      v8f acc = V8F_ZERO;
      acc = WMMA_BF16(a, bf, acc);
#pragma unroll
      for (int r = 0; r < 8; r++) S[r + 8*half][ct*16 + l16] = acc[r] * SCALE_F;
    }
  }
  __syncthreads();
  const int row = tid >> 4, c16 = tid & 15;
  float mx = -3.0e38f;
  for (int c = c16; c < LL; c += 16) mx = fmaxf(mx, S[row][c]);
  red[row][c16] = mx; __syncthreads();
  if (c16 == 0){ float m = red[row][0]; for (int i = 1; i < 16; i++) m = fmaxf(m, red[row][i]); red[row][16] = m; }
  __syncthreads();
  float m = red[row][16];
  float sm = 0.f;
  for (int c = c16; c < LL; c += 16){ float e = __expf(S[row][c] - m); S[row][c] = e; sm += e; }
  __syncthreads();
  red[row][c16] = sm; __syncthreads();
  if (c16 == 0){ float s = 0.f; for (int i = 0; i < 16; i++) s += red[row][i]; red[row][16] = s; }
  __syncthreads();
  float inv = 1.f / red[row][16];
  int l = lt*16 + row;
  if (l < LL){
    u16* prow = P + ((size_t)bh*LL + l)*320;
    for (int c = c16; c < 320; c += 16){
      u16 v = 0;
      if (c < LL) v = f2b(S[row][c] * inv);
      prow[c] = v;                                 // zero-pad keys 300..319
    }
  }
}

// ---------------- self-attention: O[b*300+l][h*32+d] = P @ V ----------------
__global__ __launch_bounds__(256) void attn_out_kernel(
    const u16* __restrict__ P, const u16* __restrict__ V, u16* __restrict__ O){
  const int bh = blockIdx.x, b = bh >> 3, hh = bh & 7;
  const int tid = threadIdx.x, wave = tid >> 5, lane = tid & 31;
  const int half = lane >> 4, l16 = lane & 15;
  const int mt = blockIdx.y*8 + wave;              // wave's 16-row M tile (0..23, 19 valid)
  __shared__ __align__(32) u16 Vt[32][32];         // transposed V tile [d][k]
  v8f acc0 = V8F_ZERO, acc1 = V8F_ZERO;
  for (int k0 = 0; k0 < 320; k0 += 32){
#pragma unroll
    for (int i = 0; i < 4; i++){
      int idx = i*256 + tid;
      int k = idx >> 5, d = idx & 31;
      int key = k0 + k; if (key > LL-1) key = LL-1; // clamped; P==0 there anyway
      Vt[d][k] = V[(size_t)(b*LL + key)*EE + hh*32 + d];
    }
    __syncthreads();
    if (mt < 19){
      int l = mt*16 + l16; if (l > LL-1) l = LL-1;
      const u16* prow = P + ((size_t)bh*LL + l)*320 + k0;
      v16bf a  = make_a_frag(prow, half);
      v16bf b0 = make_b_frag(&Vt[l16][0], half);
      v16bf b1 = make_b_frag(&Vt[16 + l16][0], half);
      acc0 = WMMA_BF16(a, b0, acc0);
      acc1 = WMMA_BF16(a, b1, acc1);
    }
    __syncthreads();
  }
  if (mt < 19){
#pragma unroll
    for (int r = 0; r < 8; r++){
      int l = mt*16 + r + 8*half;
      if (l < LL){
        u16* orow = O + (size_t)(b*LL + l)*EE + hh*32;
        orow[l16]      = f2b(acc0[r]);
        orow[16 + l16] = f2b(acc1[r]);
      }
    }
  }
}

// ---------------- cross-attn scores: softmax(mask(scale * q k^T)) over 64 keys ----------------
__global__ __launch_bounds__(256) void cross_scores_kernel(
    const u16* __restrict__ Q, const u16* __restrict__ Kmat,
    const unsigned char* __restrict__ mask, int mstride, u16* __restrict__ Aout){
  const int bh = blockIdx.x, b = bh >> 3, hh = bh & 7;
  const int lt0 = blockIdx.y*32;                   // 32 query rows per block
  const int tid = threadIdx.x, wave = tid >> 5, lane = tid & 31;
  const int half = lane >> 4, l16 = lane & 15;
  const int wm = wave >> 2, wct = wave & 3;        // 2 row-groups x 4 key-tiles
  __shared__ __align__(16) float S[32][68];
  __shared__ float red[32][9];
  {
    int l = lt0 + wm*16 + l16; if (l > LL-1) l = LL-1;
    const u16* qrow = Q + (size_t)(b*LL + l)*EE + hh*32;
    v16bf a = make_a_frag(qrow, half);
    const u16* krow = Kmat + (size_t)(b*64 + wct*16 + l16)*EE + hh*32;
    v16bf bf = make_b_frag(krow, half);
    v8f acc = V8F_ZERO;
    acc = WMMA_BF16(a, bf, acc);
#pragma unroll
    for (int r = 0; r < 8; r++){
      int mrow = wm*16 + r + 8*half;
      int c = wct*16 + l16;
      float v = acc[r] * SCALE_F;
      if (mask[(size_t)b*4096 + (size_t)c*mstride]) v = -3.0e38f;
      S[mrow][c] = v;
    }
  }
  __syncthreads();
  const int row = tid >> 3, c8 = tid & 7;
  float mx = -3.0e38f;
#pragma unroll
  for (int i = 0; i < 8; i++) mx = fmaxf(mx, S[row][c8 + 8*i]);
  red[row][c8] = mx; __syncthreads();
  if (c8 == 0){ float m = red[row][0]; for (int i = 1; i < 8; i++) m = fmaxf(m, red[row][i]); red[row][8] = m; }
  __syncthreads();
  float m = red[row][8];
  float sm = 0.f;
#pragma unroll
  for (int i = 0; i < 8; i++){ int c = c8 + 8*i; float e = __expf(S[row][c] - m); S[row][c] = e; sm += e; }
  __syncthreads();
  red[row][c8] = sm; __syncthreads();
  if (c8 == 0){ float s = 0.f; for (int i = 0; i < 8; i++) s += red[row][i]; red[row][8] = s; }
  __syncthreads();
  float inv = 1.f / red[row][8];
  int l = lt0 + row;
  if (l < LL){
    u16* arow = Aout + ((size_t)bh*LL + l)*64;
#pragma unroll
    for (int i = 0; i < 8; i++){ int c = c8 + 8*i; arow[c] = f2b(S[row][c] * inv); }
  }
}

// ---------------- fused joint cross-attention ----------------
// ca[l,d] = sum_{y,x} a_col[l,y]*a_row[l,x]*vv[(b,y,x), h*32+d]
// = GEMM per (b,h): A2[304,4096] (rank-1 on the fly) @ VV[4096,32], K order k = y*64+x.
// B tiles come from the transposed vv layout (b, e, y*64+x) so they are contiguous in K
// and staged via async global->LDS copies, double-buffered on ASYNCcnt.
__global__ __launch_bounds__(256) void joint_ca_kernel(
    const u16* __restrict__ Arow, const u16* __restrict__ Acol,
    const u16* __restrict__ VVT, u16* __restrict__ Out){
  const int bh = blockIdx.x, b = bh >> 3, hh = bh & 7;
  const int tid = threadIdx.x, wave = tid >> 5, lane = tid & 31;
  const int half = lane >> 4, l16 = lane & 15;
  const int mt = blockIdx.y*8 + wave;
  __shared__ __align__(32) u16 Bt[2][32][32];      // vv tile [d][k], double buffered
  __shared__ __align__(4)  u16 ArS[128][64];       // a_row rows of this block
  __shared__ __align__(4)  u16 AcS[128][64];       // a_col rows
#pragma unroll
  for (int i = 0; i < 32; i++){
    int idx = i*256 + tid;
    int rr = idx >> 6, c = idx & 63;
    int l = blockIdx.y*128 + rr;
    u16 va = 0, vc = 0;
    if (l < LL){
      size_t o = ((size_t)bh*LL + l)*64 + c;
      va = Arow[o]; vc = Acol[o];
    }
    ArS[rr][c] = va; AcS[rr][c] = vc;
  }
  // async staging setup: waves 0..3 copy the 32x32 B tile (one b128 per thread)
  const int sd = tid >> 2, skseg = tid & 3;
  const u16* gB = VVT + ((size_t)b*256 + hh*32 + sd)*4096 + skseg*8;
  const u32 lb[2] = { lds_off(&Bt[0][sd][skseg*8]), lds_off(&Bt[1][sd][skseg*8]) };
  if (tid < 128) async_b128(lb[0], gB);
  wait_async0();
  __syncthreads();

  v8f acc0 = V8F_ZERO, acc1 = V8F_ZERO;
  const int rr = wave*16 + l16;                    // this lane's local query row
  int cur = 0;
  for (int k0 = 0; k0 < 4096; k0 += 32){
    int nxt = cur ^ 1;
    if (k0 + 32 < 4096 && tid < 128)
      async_b128(lb[nxt], gB + k0 + 32);
    if (mt < 19){
      v16u au;
#pragma unroll
      for (int j = 0; j < 16; j++){
        int kk = k0 + 8*half + j + ((j >= 8) ? 8 : 0);   // A-fragment K mapping
        int y = kk >> 6, x = kk & 63;
        au[j] = f2b(b2f(AcS[rr][y]) * b2f(ArS[rr][x]));
      }
      v16bf a  = __builtin_bit_cast(v16bf, au);
      v16bf b0 = make_b_frag(&Bt[cur][l16][0], half);
      v16bf b1 = make_b_frag(&Bt[cur][16 + l16][0], half);
      acc0 = WMMA_BF16(a, b0, acc0);
      acc1 = WMMA_BF16(a, b1, acc1);
    }
    wait_async0();
    __syncthreads();
    cur = nxt;
  }
  if (mt < 19){
#pragma unroll
    for (int r = 0; r < 8; r++){
      int l = mt*16 + r + 8*half;
      if (l < LL){
        u16* orow = Out + (size_t)(b*LL + l)*EE + hh*32;
        orow[l16]      = f2b(acc0[r]);
        orow[16 + l16] = f2b(acc1[r]);
      }
    }
  }
}

// ---------------- host-side orchestration ----------------
extern "C" void kernel_launch(void* const* d_in, const int* in_sizes, int n_in,
                              void* d_out, int out_size, void* d_ws, size_t ws_size,
                              hipStream_t stream){
  (void)in_sizes; (void)n_in; (void)out_size; (void)ws_size;
  const float* tgt   = (const float*)d_in[0];
  const float* qpos  = (const float*)d_in[1];
  const float* qpx   = (const float*)d_in[2];
  const float* qpy   = (const float*)d_in[3];
  const float* srcs  = (const float*)d_in[4];
  const unsigned char* maskp = (const unsigned char*)d_in[5];
  const float* pemr  = (const float*)d_in[6];
  const float* pemc  = (const float*)d_in[7];
  const float* sa_w  = (const float*)d_in[8];
  const float* sa_b  = (const float*)d_in[9];
  const float* sa_ow = (const float*)d_in[10];
  const float* sa_ob = (const float*)d_in[11];
  const float* ca_w  = (const float*)d_in[12];
  const float* ca_b  = (const float*)d_in[13];
  const float* ca_ow = (const float*)d_in[14];
  const float* ca_ob = (const float*)d_in[15];
  const float* n1w   = (const float*)d_in[16];
  const float* n1b   = (const float*)d_in[17];
  const float* n2w   = (const float*)d_in[18];
  const float* n2b   = (const float*)d_in[19];
  const float* l1w   = (const float*)d_in[20];
  const float* l1b   = (const float*)d_in[21];
  const float* l2w   = (const float*)d_in[22];
  const float* l2b   = (const float*)d_in[23];
  const float* nfw   = (const float*)d_in[24];
  const float* nfb   = (const float*)d_in[25];

  // workspace carve (256B aligned)
  char* wp = (char*)d_ws;
  auto alloc = [&](size_t bytes)->void*{
    void* p = (void*)wp; wp += (bytes + 255) & ~(size_t)255; return p;
  };
  u16*  w_sa    = (u16*) alloc((size_t)768*EE*2);
  u16*  w_saow  = (u16*) alloc((size_t)EE*EE*2);
  u16*  w_ca    = (u16*) alloc((size_t)1280*EE*2);
  u16*  w_caow  = (u16*) alloc((size_t)EE*EE*2);
  u16*  w_l1    = (u16*) alloc((size_t)FFD*EE*2);
  u16*  w_l2    = (u16*) alloc((size_t)EE*FFD*2);
  u16*  qk_bf   = (u16*) alloc((size_t)NPAD*EE*2);
  u16*  tgt_bf  = (u16*) alloc((size_t)NPAD*EE*2);
  u16*  qkout   = (u16*) alloc((size_t)NPAD*512*2);
  u16*  v_bf    = (u16*) alloc((size_t)NPAD*EE*2);
  u16*  Pbuf    = (u16*) alloc((size_t)BB*HH*LL*320*2);
  u16*  o_bf    = (u16*) alloc((size_t)NPAD*EE*2);
  float* sa_f   = (float*)alloc((size_t)NPAD*EE*4);
  float* tgt2   = (float*)alloc((size_t)NPAD*EE*4);
  u16*  qrow_in = (u16*) alloc((size_t)NPAD*EE*2);
  u16*  qcol_in = (u16*) alloc((size_t)NPAD*EE*2);
  u16*  qrow    = (u16*) alloc((size_t)NPAD*EE*2);
  u16*  qcol    = (u16*) alloc((size_t)NPAD*EE*2);
  u16*  krow_in = (u16*) alloc((size_t)1024*EE*2);
  u16*  kcol_in = (u16*) alloc((size_t)1024*EE*2);
  u16*  krow    = (u16*) alloc((size_t)1024*EE*2);
  u16*  kcol    = (u16*) alloc((size_t)1024*EE*2);
  u16*  src_bf  = (u16*) alloc((size_t)BB*4096*EE*2);
  u16*  vvT     = (u16*) alloc((size_t)BB*EE*4096*2);   // (b, e, y*64+x)
  u16*  arow    = (u16*) alloc((size_t)BB*HH*LL*64*2);
  u16*  acol    = (u16*) alloc((size_t)BB*HH*LL*64*2);
  u16*  capre   = (u16*) alloc((size_t)NPAD*EE*2);
  float* caproj = (float*)alloc((size_t)NPAD*EE*4);
  float* tgt3   = (float*)alloc((size_t)NPAD*EE*4);
  u16*  tgt3_bf = (u16*) alloc((size_t)NPAD*EE*2);
  u16*  ff1     = (u16*) alloc((size_t)NPAD*FFD*2);
  float* ff2    = (float*)alloc((size_t)NPAD*EE*4);

  const int NE = NTOK*EE;
  // ---- weights / activations -> bf16 ----
  cvtbf_kernel<<<768, 256, 0, stream>>>(sa_w,  w_sa,   768*EE);
  cvtbf_kernel<<<256, 256, 0, stream>>>(sa_ow, w_saow, EE*EE);
  cvtbf_kernel<<<1280,256, 0, stream>>>(ca_w,  w_ca,   1280*EE);
  cvtbf_kernel<<<256, 256, 0, stream>>>(ca_ow, w_caow, EE*EE);
  cvtbf_kernel<<<1024,256, 0, stream>>>(l1w,   w_l1,   FFD*EE);
  cvtbf_kernel<<<1024,256, 0, stream>>>(l2w,   w_l2,   EE*FFD);
  add2bf_kernel<<<NTOK,256, 0, stream>>>(tgt, qpos, qk_bf, NE);
  cvtbf_kernel<<<NTOK, 256, 0, stream>>>(tgt, tgt_bf, NE);

  // ---- self-attention ----
  gemm_bf16_kernel<<<dim3(NPAD/64, 512/64), 256, 0, stream>>>(qk_bf, w_sa, sa_b, nullptr, qkout, nullptr, 512, EE, 0);
  gemm_bf16_kernel<<<dim3(NPAD/64, EE/64),  256, 0, stream>>>(tgt_bf, w_sa + 512*EE, sa_b + 512, nullptr, v_bf, nullptr, EE, EE, 0);
  attn_scores_kernel<<<dim3(BB*HH, 19), 256, 0, stream>>>(qkout, Pbuf);
  attn_out_kernel<<<dim3(BB*HH, 3), 256, 0, stream>>>(Pbuf, v_bf, o_bf);
  gemm_bf16_kernel<<<dim3(NPAD/64, EE/64), 256, 0, stream>>>(o_bf, w_saow, sa_ob, sa_f, nullptr, nullptr, EE, EE, 0);
  add_ln_kernel<<<NTOK/8, 256, 0, stream>>>(tgt, sa_f, n2w, n2b, tgt2, nullptr, NTOK);

  // ---- cross-attention inputs ----
  add2bf_kernel<<<NTOK, 256, 0, stream>>>(tgt2, qpx, qrow_in, NE);
  add2bf_kernel<<<NTOK, 256, 0, stream>>>(tgt2, qpy, qcol_in, NE);
  make_kin_kernel<<<1024, 256, 0, stream>>>(srcs, pemr, krow_in, 0);
  make_kin_kernel<<<1024, 256, 0, stream>>>(srcs, pemc, kcol_in, 1);
  src_tr_kernel<<<65536, 256, 0, stream>>>(srcs, src_bf);

  gemm_bf16_kernel<<<dim3(NPAD/64, EE/64), 256, 0, stream>>>(qrow_in, w_ca,            ca_b,        nullptr, qrow, nullptr, EE, EE, 0);
  gemm_bf16_kernel<<<dim3(NPAD/64, EE/64), 256, 0, stream>>>(qcol_in, w_ca + 1*EE*EE,  ca_b + 256,  nullptr, qcol, nullptr, EE, EE, 0);
  gemm_bf16_kernel<<<dim3(1024/64, EE/64), 256, 0, stream>>>(krow_in, w_ca + 2*EE*EE,  ca_b + 512,  nullptr, krow, nullptr, EE, EE, 0);
  gemm_bf16_kernel<<<dim3(1024/64, EE/64), 256, 0, stream>>>(kcol_in, w_ca + 3*EE*EE,  ca_b + 768,  nullptr, kcol, nullptr, EE, EE, 0);
  gemm_bf16_kernel<<<dim3(65536/64, EE/64),256, 0, stream>>>(src_bf,  w_ca + 4*EE*EE,  ca_b + 1024, nullptr, nullptr, vvT, EE, EE, 0);

  cross_scores_kernel<<<dim3(BB*HH, 10), 256, 0, stream>>>(qrow, krow, maskp, 1,  arow);
  cross_scores_kernel<<<dim3(BB*HH, 10), 256, 0, stream>>>(qcol, kcol, maskp, 64, acol);
  joint_ca_kernel<<<dim3(BB*HH, 3), 256, 0, stream>>>(arow, acol, vvT, capre);
  gemm_bf16_kernel<<<dim3(NPAD/64, EE/64), 256, 0, stream>>>(capre, w_caow, ca_ob, caproj, nullptr, nullptr, EE, EE, 0);
  add_ln_kernel<<<NTOK/8, 256, 0, stream>>>(tgt2, caproj, n1w, n1b, tgt3, tgt3_bf, NTOK);

  // ---- FFN ----
  gemm_bf16_kernel<<<dim3(NPAD/64, FFD/64), 256, 0, stream>>>(tgt3_bf, w_l1, l1b, nullptr, ff1, nullptr, FFD, EE, 1);
  gemm_bf16_kernel<<<dim3(NPAD/64, EE/64),  256, 0, stream>>>(ff1, w_l2, l2b, ff2, nullptr, nullptr, EE, FFD, 0);
  add_ln_kernel<<<NTOK/8, 256, 0, stream>>>(tgt3, ff2, nfw, nfb, (float*)d_out, nullptr, NTOK);
}
